// SingleHeadAttention_6485400617440
// MI455X (gfx1250) — matrix operand order repaired and checked by reading, not compiled, not run on hardware
//
#include <hip/hip_runtime.h>
#include <hip/hip_bf16.h>

// ---------------------------------------------------------------------------
// Single-head causal attention for MI455X (gfx1250), wave32 + WMMA bf16 +
// Tensor Data Mover staging of K/V tiles into LDS.
// B=4, S=4096, D=1024, H=64. fp32 in/out, bf16 matrix operands, f32 accum.
// ---------------------------------------------------------------------------

typedef __bf16 bf16_t;
typedef __attribute__((ext_vector_type(16))) __bf16 v16bf;
typedef __attribute__((ext_vector_type(8)))  __bf16 v8bf;
typedef __attribute__((ext_vector_type(8)))  float  v8f;
typedef __attribute__((ext_vector_type(4)))  float  v4f;
typedef __attribute__((ext_vector_type(4)))  unsigned v4u;
typedef __attribute__((ext_vector_type(8)))  int    v8i;
typedef __attribute__((ext_vector_type(4)))  int    v4i;

#define B_   4
#define S_   4096
#define D_   1024
#define H_   64
#define MTOT (B_ * S_)          // 16384 rows total

#if __has_builtin(__builtin_amdgcn_tensor_load_to_lds) && \
    __has_builtin(__builtin_amdgcn_s_wait_tensorcnt)
#define HAVE_TDM 1
#else
#define HAVE_TDM 0
#endif

__device__ __forceinline__ bf16_t f2bf(float f) { return (bf16_t)f; }

__device__ __forceinline__ v16bf cat8(v8bf lo, v8bf hi) {
  return __builtin_shufflevector(lo, hi, 0, 1, 2, 3, 4, 5, 6, 7,
                                         8, 9, 10, 11, 12, 13, 14, 15);
}

// ---------------------------------------------------------------------------
// Kernel 1: transpose + convert the three weight matrices to bf16.
// wT layout: [o][H][D], o = 0:K, 1:Q, 2:V  (column-contiguous for B-fragments)
// ---------------------------------------------------------------------------
__global__ void wcvt_kernel(const float* __restrict__ Wk,
                            const float* __restrict__ Wq,
                            const float* __restrict__ Wv,
                            bf16_t* __restrict__ wT) {
  int idx = blockIdx.x * blockDim.x + threadIdx.x;       // over H*D
  if (idx >= H_ * D_) return;
  int h = idx / D_;
  int d = idx - h * D_;
  wT[0 * H_ * D_ + idx] = f2bf(Wk[d * H_ + h]);
  wT[1 * H_ * D_ + idx] = f2bf(Wq[d * H_ + h]);
  wT[2 * H_ * D_ + idx] = f2bf(Wv[d * H_ + h]);
}

// ---------------------------------------------------------------------------
// Kernel 2: QKV projection.  One wave = one 32-row M tile (2 A-fragments
// sharing every B-fragment), N=64 (4 n-tiles), K=1024 in steps of 32.
// 24 bf16 WMMAs per K step vs 8 A-loads + 24 B-loads.  Bias folded in
// epilogue; softmax scale (1/8) folded into q.  q,k stored row-major bf16
// [MTOT,H]; v stored transposed bf16 [B][H][S] for contiguous PV B-fragments.
// ---------------------------------------------------------------------------
__global__ __launch_bounds__(256) void qkv_kernel(
    const float* __restrict__ x, const bf16_t* __restrict__ wT,
    const float* __restrict__ bk, const float* __restrict__ bq,
    const float* __restrict__ bv,
    bf16_t* __restrict__ qb, bf16_t* __restrict__ kb, bf16_t* __restrict__ vT) {
  const int lane = threadIdx.x & 31;
  const int l16  = lane & 15;
  const int half = lane >> 4;
  const int wave = threadIdx.x >> 5;
  const int m0   = (blockIdx.x * 8 + wave) * 32;

  v8f zero = {};
  v8f acc[3][4][2];
  #pragma unroll
  for (int o = 0; o < 3; ++o)
    #pragma unroll
    for (int n = 0; n < 4; ++n)
      #pragma unroll
      for (int mm = 0; mm < 2; ++mm) acc[o][n][mm] = zero;

  const float* xrow[2] = {x + (size_t)(m0 + l16) * D_,
                          x + (size_t)(m0 + 16 + l16) * D_};

  for (int k0 = 0; k0 < D_; k0 += 32) {
    // A-fragments (16x32 bf16): lanes 0-15 K={k0..k0+7, k0+16..k0+23},
    // lanes 16-31 shifted by 8.  Convert fp32 x on the fly (native pk cvt).
    const int ka = k0 + 8 * half;
    v16bf A[2];
    #pragma unroll
    for (int mm = 0; mm < 2; ++mm) {
      v4f a0 = *(const v4f*)(xrow[mm] + ka);
      v4f a1 = *(const v4f*)(xrow[mm] + ka + 4);
      v4f a2 = *(const v4f*)(xrow[mm] + ka + 16);
      v4f a3 = *(const v4f*)(xrow[mm] + ka + 20);
      v8bf lo, hi;
      #pragma unroll
      for (int i = 0; i < 4; ++i) {
        lo[i]     = f2bf(a0[i]);
        lo[4 + i] = f2bf(a1[i]);
        hi[i]     = f2bf(a2[i]);
        hi[4 + i] = f2bf(a3[i]);
      }
      A[mm] = cat8(lo, hi);
    }
    #pragma unroll
    for (int o = 0; o < 3; ++o) {
      #pragma unroll
      for (int n = 0; n < 4; ++n) {
        // B-fragment (32x16): lane = output column, K = {8h..8h+7, 16+8h..16+8h+7};
        // reused by both A-fragments.
        const bf16_t* wrow =
            wT + ((size_t)o * H_ + n * 16 + l16) * D_ + k0 + 8 * half;
        v16bf Bm = cat8(*(const v8bf*)(wrow), *(const v8bf*)(wrow + 16));
        #pragma unroll
        for (int mm = 0; mm < 2; ++mm)
          acc[o][n][mm] = __builtin_amdgcn_wmma_f32_16x16x32_bf16(
              false, A[mm], false, Bm, (short)0, acc[o][n][mm], false, false);
      }
    }
  }

  // Epilogue: bias add + store.  D-fragment: lane -> col = l16, rows r+8*half.
  const int batch = m0 / S_;
  const int srow  = m0 - batch * S_;
  #pragma unroll
  for (int n = 0; n < 4; ++n) {
    const int col = n * 16 + l16;
    const float biask = bk[col];
    const float biasq = bq[col];
    const float biasv = bv[col];
    #pragma unroll
    for (int mm = 0; mm < 2; ++mm) {
      const int mb = m0 + mm * 16;
      #pragma unroll
      for (int r = 0; r < 8; ++r)
        kb[(size_t)(mb + r + 8 * half) * H_ + col] =
            f2bf(acc[0][n][mm][r] + biask);
      #pragma unroll
      for (int r = 0; r < 8; ++r)
        qb[(size_t)(mb + r + 8 * half) * H_ + col] =
            f2bf((acc[1][n][mm][r] + biasq) * 0.125f);
      v8bf vv;
      #pragma unroll
      for (int r = 0; r < 8; ++r) vv[r] = f2bf(acc[2][n][mm][r] + biasv);
      *(v8bf*)(vT + ((size_t)batch * H_ + col) * S_ + srow + mm * 16 +
               8 * half) = vv;
    }
  }
}

// ---------------------------------------------------------------------------
// TDM 2D tile load: builds the D# per CDNA5 ISA §8 (groups 0/1; groups 2/3
// zero for <=2D) and issues tensor_load_to_lds (6-arg toolchain form).
// Data size = 2 bytes (bf16).  padctl carries pad_enable/pad_interval/
// pad_amount bits for group1 dword0.
// ---------------------------------------------------------------------------
#if HAVE_TDM
__device__ __forceinline__ void tdm_load_2d(unsigned lds_off, const void* gptr,
                                            unsigned td0, unsigned td1,
                                            unsigned tl0, unsigned tl1,
                                            unsigned stride0, unsigned padctl) {
  unsigned long long ga = (unsigned long long)gptr;
  v4u g0;
  g0[0] = 1u;                                    // count=1 (valid), user mode
  g0[1] = lds_off;                               // lds_addr
  g0[2] = (unsigned)ga;                          // global_addr[31:0]
  g0[3] = (unsigned)(ga >> 32) | (2u << 30);     // global_addr[56:32], type=2
  v8i g1;
  g1[0] = (int)((1u << 16) | padctl);            // data_size=2B + pad ctl
  g1[1] = (int)((td0 & 0xffffu) << 16);          // tensor_dim0[15:0]
  g1[2] = (int)((td0 >> 16) | ((td1 & 0xffffu) << 16));
  g1[3] = (int)((td1 >> 16) | (tl0 << 16));      // tile_dim0
  g1[4] = (int)tl1;                              // tile_dim1 (tile_dim2=0)
  g1[5] = (int)stride0;                          // tensor_dim0_stride[31:0]
  g1[6] = 0;
  g1[7] = 0;
  v4i z4 = {};
  v8i z8 = {};
  __builtin_amdgcn_tensor_load_to_lds(g0, g1, z4, z4, z8, 0);
}
#endif

// LDS row strides (elements) after TDM padding -> conflict-free ds_load_b128
#define KSTRIDE 72   // 64 elems (128B) + 4-dword pad = 36 dwords
#define VSTRIDE 40   // 32 elems (64B)  + 4-dword pad = 20 dwords

__device__ __forceinline__ void stage_tiles(bf16_t* lk, bf16_t* lv,
                                            const bf16_t* kbase,
                                            const bf16_t* vbase, int n0,
                                            int tid, int wave) {
#if HAVE_TDM
  if (wave == 0) {
    // K tile: 32 rows x 64 h from row-major [S,H]; pad 4 dwords per 128B row
    tdm_load_2d((unsigned)(unsigned long long)lk, kbase + (size_t)n0 * H_,
                H_, S_, H_, 32, H_, (1u << 20) | (4u << 22) | (3u << 25));
    // V tile: 64 rows x 32 s from row-major [H,S]; pad 4 dwords per 64B row
    tdm_load_2d((unsigned)(unsigned long long)lv, vbase + n0,
                S_, H_, 32, H_, S_, (1u << 20) | (3u << 22) | (3u << 25));
  }
#else
  {
    int r = tid >> 2, c = (tid & 3) * 16;        // 32 rows x 64 elems
    const bf16_t* src = kbase + (size_t)(n0 + r) * H_ + c;
    *(v8bf*)&lk[r * KSTRIDE + c]     = *(const v8bf*)src;
    *(v8bf*)&lk[r * KSTRIDE + c + 8] = *(const v8bf*)(src + 8);
    int vr = tid >> 1, vc = (tid & 1) * 16;      // 64 rows x 32 elems
    const bf16_t* vsrc = vbase + (size_t)vr * S_ + n0 + vc;
    *(v8bf*)&lv[vr * VSTRIDE + vc]     = *(const v8bf*)vsrc;
    *(v8bf*)&lv[vr * VSTRIDE + vc + 8] = *(const v8bf*)(vsrc + 8);
  }
#endif
}

// ---------------------------------------------------------------------------
// Kernel 3: flash attention.  One block (4 waves) owns a 64-query tile; each
// wave one 16-query subtile.  K/V 32-key tiles are double-buffered in LDS via
// TDM.  Computes S^T = K_blk @ Q^T so the f32 D layout (lane=query, regs=key)
// is exactly the 16-bit A-fragment layout needed for P@V -> no transpose.
// ---------------------------------------------------------------------------
__global__ __launch_bounds__(128) void attn_kernel(
    const bf16_t* __restrict__ qb, const bf16_t* __restrict__ kb,
    const bf16_t* __restrict__ vT, float* __restrict__ out) {
  __shared__ bf16_t ldsK[2][32 * KSTRIDE];   // 2 x 4.5 KB
  __shared__ bf16_t ldsV[2][64 * VSTRIDE];   // 2 x 5.0 KB

  const int lane  = threadIdx.x & 31;
  const int l16   = lane & 15;
  const int half  = lane >> 4;
  const int wave  = threadIdx.x >> 5;

  const int blk64 = blockIdx.x;              // 64 blocks launched, 64 queries each
  const int batch = blk64 >> 6;              // 64 blocks per batch
  const int mq0b  = (blk64 & 63) << 6;       // block's query base in batch
  const int mq0   = mq0b + wave * 16;        // this wave's query base
  const size_t gq = (size_t)batch * S_;

  const bf16_t* kbase = kb + gq * H_;
  const bf16_t* vbase = vT + (size_t)batch * H_ * S_;
  const int nblk = (mq0b >> 5) + 2;          // 32-key blocks for whole block

  // Q^T B-fragments (loaded once): lane = query column, 16 consecutive h.
  const bf16_t* qrow = qb + (gq + mq0 + l16) * H_;
  v16bf Bq[2];
  #pragma unroll
  for (int j = 0; j < 2; ++j)
    Bq[j] = cat8(*(const v8bf*)(qrow + j * 32 + 16 * half),
                 *(const v8bf*)(qrow + j * 32 + 16 * half + 8));

  v8f zero = {};
  v8f O[4];
  #pragma unroll
  for (int h = 0; h < 4; ++h) O[h] = zero;
  float rmax = -3.0e38f;
  float rsum = 0.0f;

  stage_tiles(&ldsK[0][0], &ldsV[0][0], kbase, vbase, 0, threadIdx.x, wave);

  for (int blk = 0; blk < nblk; ++blk) {
    const int n0  = blk * 32;
    const int buf = blk & 1;
#if HAVE_TDM
    if (wave == 0) __builtin_amdgcn_s_wait_tensorcnt(0);
#endif
    __syncthreads();                 // tiles for blk visible; other buf free
    if (blk + 1 < nblk)
      stage_tiles(&ldsK[buf ^ 1][0], &ldsV[buf ^ 1][0], kbase, vbase, n0 + 32,
                  threadIdx.x, wave);

    // ---- S^T tiles: 16(key) x 16(query), contraction over h=64 ----
    v8f St[2];
    #pragma unroll
    for (int t = 0; t < 2; ++t) {
      v8f acc = zero;
      const bf16_t* kr = &ldsK[buf][(t * 16 + l16) * KSTRIDE];
      #pragma unroll
      for (int j = 0; j < 2; ++j) {
        v16bf A = cat8(*(const v8bf*)(kr + j * 32 + 16 * half),
                       *(const v8bf*)(kr + j * 32 + 16 * half + 8));
        acc = __builtin_amdgcn_wmma_f32_16x16x32_bf16(
            false, A, false, Bq[j], (short)0, acc, false, false);
      }
      St[t] = acc;
    }

    // ---- causal mask (diagonal / beyond-diagonal blocks per wave) ----
    if (n0 + 31 > mq0) {
      const int m = mq0 + l16;
      #pragma unroll
      for (int t = 0; t < 2; ++t)
        #pragma unroll
        for (int r = 0; r < 8; ++r)
          if (n0 + t * 16 + r + 8 * half > m) St[t][r] = -3.0e38f;
    }

    // ---- online softmax over query column m = mq0 + l16 ----
    float bmax = -3.0e38f;
    #pragma unroll
    for (int t = 0; t < 2; ++t)
      #pragma unroll
      for (int r = 0; r < 8; ++r) bmax = fmaxf(bmax, St[t][r]);
    bmax = fmaxf(bmax, __shfl_xor(bmax, 16, 32));
    const float mnew = fmaxf(rmax, bmax);
    const float corr = __expf(rmax - mnew);

    float psum = 0.0f;
    v16bf P;                                   // already A-fragment layout
    #pragma unroll
    for (int t = 0; t < 2; ++t)
      #pragma unroll
      for (int r = 0; r < 8; ++r) {
        float p = __expf(St[t][r] - mnew);
        psum += p;
        P[t * 8 + r] = f2bf(p);
      }
    psum += __shfl_xor(psum, 16, 32);
    rsum = rsum * corr + psum;
    rmax = mnew;

    // ---- rescale O (broadcast corr per O-row m = r + 8*half) ----
    #pragma unroll
    for (int r = 0; r < 8; ++r) {
      float c = __shfl(corr, r + 8 * half, 32);
      #pragma unroll
      for (int h = 0; h < 4; ++h) O[h][r] *= c;
    }

    // ---- O += P @ V  (B-fragment: keys {8h..8h+7, 16+8h..16+8h+7} from padded LDS) ----
    #pragma unroll
    for (int h = 0; h < 4; ++h) {
      const bf16_t* vr = &ldsV[buf][(h * 16 + l16) * VSTRIDE + 8 * half];
      v16bf Bv = cat8(*(const v8bf*)vr, *(const v8bf*)(vr + 16));
      O[h] = __builtin_amdgcn_wmma_f32_16x16x32_bf16(
          false, P, false, Bv, (short)0, O[h], false, false);
    }
  }

  // ---- epilogue: divide by row sum, store fp32 ----
  const float linv = 1.0f / rsum;
  #pragma unroll
  for (int r = 0; r < 8; ++r) {
    const float li = __shfl(linv, r + 8 * half, 32);
    const size_t row = gq + mq0 + r + 8 * half;
    #pragma unroll
    for (int h = 0; h < 4; ++h)
      out[row * H_ + h * 16 + l16] = O[h][r] * li;
  }
}

// ---------------------------------------------------------------------------
extern "C" void kernel_launch(void* const* d_in, const int* in_sizes, int n_in,
                              void* d_out, int out_size, void* d_ws,
                              size_t ws_size, hipStream_t stream) {
  const float* x  = (const float*)d_in[0];
  const float* Wk = (const float*)d_in[1];
  const float* bk = (const float*)d_in[2];
  const float* Wq = (const float*)d_in[3];
  const float* bq = (const float*)d_in[4];
  const float* Wv = (const float*)d_in[5];
  const float* bv = (const float*)d_in[6];

  char* ws = (char*)d_ws;
  bf16_t* wT = (bf16_t*)(ws);                        // 3*64*1024*2 = 384 KB
  bf16_t* qb = (bf16_t*)(ws + 0x80000);              // 16384*64*2  = 2 MB
  bf16_t* kb = (bf16_t*)(ws + 0x80000 + 0x200000);   // 2 MB
  bf16_t* vT = (bf16_t*)(ws + 0x80000 + 0x400000);   // 2 MB  (~6.5 MB total)

  wcvt_kernel<<<dim3((H_ * D_ + 255) / 256), dim3(256), 0, stream>>>(
      Wk, Wq, Wv, wT);
  qkv_kernel<<<dim3(MTOT / 32 / 8), dim3(256), 0, stream>>>(
      x, wT, bk, bq, bv, qb, kb, vT);
  attn_kernel<<<dim3(MTOT / 16 / 4 / 4), dim3(128), 0, stream>>>(
      qb, kb, vT, (float*)d_out);
}
